// DownSamplingSpatial2ChannelFVDB_21345987461766
// MI455X (gfx1250) — matrix operand order repaired and checked by reading, not compiled, not run on hardware
//
#include <hip/hip_runtime.h>
#include <hip/hip_bf16.h>

// ---------------------------------------------------------------------------
// DownSamplingSpatial2ChannelFVDB on MI455X (gfx1250, wave32, WMMA)
//
// Fused single-GEMM formulation:
//   out (M x 256) = Gather(in_data) (M x 256)  @  W_big (256 x 256)
//   W_big[s*32+i, o] = sum_j W_mid[i,j] * W_out[s*32+j, o]
// Main GEMM runs as bf16x3 split-precision WMMA (fp32 accumulate).
// ---------------------------------------------------------------------------

typedef __attribute__((ext_vector_type(16))) __bf16 v16bf;
typedef __attribute__((ext_vector_type(8)))  float  v8f;

__device__ __forceinline__ unsigned short f2bf(float f) {
    unsigned u = __float_as_uint(f);
    u += 0x7fffu + ((u >> 16) & 1u);          // round-to-nearest-even
    return (unsigned short)(u >> 16);
}
__device__ __forceinline__ float bf2f(unsigned short h) {
    return __uint_as_float(((unsigned)h) << 16);
}

// ---- Pass 1: W_big[r, o] = sum_j W_mid[i,j] * W_out[s*32+j, o], r = s*32+i
__global__ void k_wbig(const float* __restrict__ Wmid,
                       const float* __restrict__ Wout,
                       float* __restrict__ Wbig) {
    int t = blockIdx.x * blockDim.x + threadIdx.x;   // 0 .. 65535
    int o = t & 255;
    int r = t >> 8;                                   // 0 .. 255
    int s = r >> 5, i = r & 31;
    float acc = 0.f;
#pragma unroll
    for (int j = 0; j < 32; ++j)
        acc += Wmid[i * 32 + j] * Wout[(s * 32 + j) * 256 + o];
    Wbig[r * 256 + o] = acc;
}

// ---- Pass 2: repack W_big into wave32 bf16 B-fragments (hi & lo halves).
// Slot index t = ((nb*8 + ks)*32 + lane)*8 + j ; each slot = packed bf16 pair
// B layout (32x16 per k-step): lane<16 -> K = ks*32 + {0..15}, N = nb*16+lane
//                              lane>=16 -> K = ks*32 + {16..31}, N = nb*16+(lane-16)
__global__ void k_pack(const float* __restrict__ Wbig,
                       unsigned* __restrict__ BH,
                       unsigned* __restrict__ BL) {
    int t = blockIdx.x * blockDim.x + threadIdx.x;   // 0 .. 32767
    int j    = t & 7;
    int lane = (t >> 3) & 31;
    int ks   = (t >> 8) & 7;
    int nb   = t >> 11;                               // 0 .. 15
    int n  = nb * 16 + (lane & 15);
    int k0 = ks * 32 + ((lane < 16) ? 0 : 16) + 2 * j;
    float f0 = Wbig[k0 * 256 + n];
    float f1 = Wbig[(k0 + 1) * 256 + n];
    unsigned short h0 = f2bf(f0), h1 = f2bf(f1);
    unsigned short l0 = f2bf(f0 - bf2f(h0));
    unsigned short l1 = f2bf(f1 - bf2f(h1));
    BH[t] = (unsigned)h0 | ((unsigned)h1 << 16);
    BL[t] = (unsigned)l0 | ((unsigned)l1 << 16);
}

// ---- Pass 3: inverse of the scatter permutation (s = 2 -> 8 children)
__global__ void k_inv(const int* __restrict__ ijk,
                      const int* __restrict__ down,
                      int* __restrict__ inv, int N) {
    int n = blockIdx.x * blockDim.x + threadIdx.x;
    if (n >= N) return;
    int l = (ijk[3 * n] & 1) * 4 + (ijk[3 * n + 1] & 1) * 2 + (ijk[3 * n + 2] & 1);
    inv[down[n] * 8 + l] = n;
}

// ---- Pass 4: main GEMM, 16x256 output tile per 512-thread block (16 waves)
#define LDSW 264   // padded bf16 row stride (256 + 8) -> conflict-free column reads

__global__ __launch_bounds__(512) void k_gemm(const float* __restrict__ in_data,
                                              const int* __restrict__ inv,
                                              const unsigned* __restrict__ BH,
                                              const unsigned* __restrict__ BL,
                                              float* __restrict__ out) {
    __shared__ __align__(16) unsigned short ldsH[16 * LDSW];
    __shared__ __align__(16) unsigned short ldsL[16 * LDSW];

    const int tid = threadIdx.x;
    const int m0  = blockIdx.x * 16;

    // Gathered A-tile load + one-time fp32 -> bf16 hi/lo split.
    // 16 rows x 256 floats; chunk = (row, s) of 32 contiguous floats; 4 threads/chunk.
    {
        int chunk = tid >> 2;                 // 0 .. 127
        int part  = tid & 3;                  // 0 .. 3
        int row = chunk >> 3;
        int s   = chunk & 7;
        int src = inv[(m0 + row) * 8 + s];
        const float4* p = (const float4*)(in_data + (long)src * 32 + part * 8);
        float4 f0 = p[0];
        float4 f1 = p[1];
        float v[8] = {f0.x, f0.y, f0.z, f0.w, f1.x, f1.y, f1.z, f1.w};
        unsigned short h[8], l[8];
#pragma unroll
        for (int q = 0; q < 8; ++q) {
            h[q] = f2bf(v[q]);
            l[q] = f2bf(v[q] - bf2f(h[q]));
        }
        int base = row * LDSW + s * 32 + part * 8;   // even -> dword aligned
        unsigned* dh = (unsigned*)&ldsH[base];
        unsigned* dl = (unsigned*)&ldsL[base];
#pragma unroll
        for (int q = 0; q < 4; ++q) {
            dh[q] = (unsigned)h[2 * q] | ((unsigned)h[2 * q + 1] << 16);
            dl[q] = (unsigned)l[2 * q] | ((unsigned)l[2 * q + 1] << 16);
        }
    }
    __syncthreads();

    const int lane = tid & 31;
    const int w    = tid >> 5;               // column block 0..15 (N0 = 16*w)

    v8f c = {0.f, 0.f, 0.f, 0.f, 0.f, 0.f, 0.f, 0.f};

    // A frag: lane<16 -> M=lane,    K = kb + {0..15}
    //         lane>=16 -> M=lane-16, K = kb + {16..31}
    const int rowIdx = lane & 15;
    const int aoff   = (lane < 16) ? 0 : 16;
    const unsigned short* rowH = &ldsH[rowIdx * LDSW];
    const unsigned short* rowL = &ldsL[rowIdx * LDSW];
    const unsigned* bh = BH + w * 2048 + lane * 8;
    const unsigned* bl = BL + w * 2048 + lane * 8;

    union AU { uint4 q[2]; v16bf v; };
    union BU { uint4 q[2]; v16bf v; };

#pragma unroll
    for (int ks = 0; ks < 8; ++ks) {
        const int kb = ks * 32;
        AU aH, aL;
        aH.q[0] = *(const uint4*)(rowH + kb + aoff);
        aH.q[1] = *(const uint4*)(rowH + kb + 8 + aoff);
        aL.q[0] = *(const uint4*)(rowL + kb + aoff);
        aL.q[1] = *(const uint4*)(rowL + kb + 8 + aoff);
        BU bHf, bLf;
        const uint4* ph = (const uint4*)(bh + ks * 256);
        const uint4* pl = (const uint4*)(bl + ks * 256);
        bHf.q[0] = ph[0]; bHf.q[1] = ph[1];
        bLf.q[0] = pl[0]; bLf.q[1] = pl[1];

        c = __builtin_amdgcn_wmma_f32_16x16x32_bf16(false, aH.v, false, bHf.v,
                                                    (short)0, c, false, false);
        c = __builtin_amdgcn_wmma_f32_16x16x32_bf16(false, aH.v, false, bLf.v,
                                                    (short)0, c, false, false);
        c = __builtin_amdgcn_wmma_f32_16x16x32_bf16(false, aL.v, false, bHf.v,
                                                    (short)0, c, false, false);
    }

    // C/D layout: VGPR v -> row (v or v+8), col = lane&15 within block w
    const int col   = w * 16 + (lane & 15);
    const int rbase = m0 + ((lane < 16) ? 0 : 8);
#pragma unroll
    for (int v = 0; v < 8; ++v)
        out[(long)(rbase + v) * 256 + col] = c[v];
}

// ---------------------------------------------------------------------------
extern "C" void kernel_launch(void* const* d_in, const int* in_sizes, int n_in,
                              void* d_out, int out_size, void* d_ws, size_t ws_size,
                              hipStream_t stream) {
    const float* in_data = (const float*)d_in[0];
    const float* Wmid    = (const float*)d_in[1];
    const float* Wout    = (const float*)d_in[2];
    const int*   ijk     = (const int*)d_in[3];
    const int*   down    = (const int*)d_in[4];

    const int N = in_sizes[0] / 32;   // fine voxels
    const int M = N / 8;              // coarse voxels

    // workspace layout
    char* ws = (char*)d_ws;
    float*    Wbig = (float*)ws;                      // 256*256*4   = 262144 B
    unsigned* BH   = (unsigned*)(ws + 262144);        // 32768*4     = 131072 B
    unsigned* BL   = (unsigned*)(ws + 393216);        // 32768*4     = 131072 B
    int*      inv  = (int*)(ws + 524288);             // N*4         = 8 MB

    k_wbig<<<256, 256, 0, stream>>>(Wmid, Wout, Wbig);
    k_pack<<<128, 256, 0, stream>>>(Wbig, BH, BL);
    k_inv<<<(N + 255) / 256, 256, 0, stream>>>(ijk, down, inv, N);
    k_gemm<<<M / 16, 512, 0, stream>>>(in_data, inv, BH, BL, (float*)d_out);
}
